// DifferentialAttention_65687229825427
// MI455X (gfx1250) — matrix operand order repaired and checked by reading, not compiled, not run on hardware
//
#include <hip/hip_runtime.h>

typedef __bf16 bf16_t;
typedef __attribute__((ext_vector_type(16))) __bf16 v16bf;
typedef __attribute__((ext_vector_type(8)))  __bf16 v8bf;
typedef __attribute__((ext_vector_type(4)))  __bf16 v4bf;
typedef __attribute__((ext_vector_type(8)))  float  v8f;
typedef __attribute__((ext_vector_type(4)))  float  v4f;
typedef __attribute__((ext_vector_type(4)))  unsigned int u32x4;
typedef __attribute__((ext_vector_type(4)))  int    i32x4;
typedef __attribute__((ext_vector_type(8)))  int    i32x8;

#define TT      2048
#define DMODEL  1024
#define NH      16
#define HD      64
#define QKVN    5120   // NH * 5 * HD
#define BB      2
#define AON     (NH*HD) // 1024

#if defined(__has_builtin)
#  if __has_builtin(__builtin_amdgcn_tensor_load_to_lds) && __has_builtin(__builtin_amdgcn_s_wait_tensorcnt)
#    define HAVE_TDM 1
#  else
#    define HAVE_TDM 0
#  endif
#else
#  define HAVE_TDM 0
#endif

static __device__ __forceinline__ v8f wmma_bf16(v16bf a, v16bf b, v8f c) {
  return __builtin_amdgcn_wmma_f32_16x16x32_bf16(false, a, false, b, (short)0, c, false, false);
}

// ---- fragment loaders (wave32 WMMA operands, contiguous k order used alike for A and B) ----
// A (16xK=32): lane r=lane&15 holds row r; elem e -> K=k0+16*half+e
static __device__ __forceinline__ v16bf load_a_f32(const float* __restrict__ src, int row0, int k0, int ld, int lane) {
  int half = lane >> 4, r = lane & 15;
  const float* p = src + (size_t)(row0 + r) * ld + k0 + 16 * half;
  v4f x0 = *(const v4f*)p;
  v4f x1 = *(const v4f*)(p + 4);
  v4f x2 = *(const v4f*)(p + 8);
  v4f x3 = *(const v4f*)(p + 12);
  v16bf a;
#pragma unroll
  for (int t = 0; t < 4; ++t) { a[t] = (bf16_t)x0[t]; a[4 + t] = (bf16_t)x1[t]; }
#pragma unroll
  for (int t = 0; t < 4; ++t) { a[8 + t] = (bf16_t)x2[t]; a[12 + t] = (bf16_t)x3[t]; }
  return a;
}
static __device__ __forceinline__ v16bf load_a_bf16(const bf16_t* src, int row0, int k0, int ld, int lane) {
  int half = lane >> 4, r = lane & 15;
  const bf16_t* p = src + (size_t)(row0 + r) * ld + k0 + 16 * half;
  v8bf lo = *(const v8bf*)p;        // 16B aligned
  v8bf hi = *(const v8bf*)(p + 8);  // 16B aligned
  v16bf a;
#pragma unroll
  for (int t = 0; t < 8; ++t) { a[t] = lo[t]; a[8 + t] = hi[t]; }
  return a;
}
// B (K=32 x 16): lane n=lane&15 holds col n; elem e -> K = k0+16*half+e. src row-major [K, N].
static __device__ __forceinline__ v16bf load_b_bf16(const bf16_t* src, int k0, int n0, int ld, int lane) {
  int half = lane >> 4, n = lane & 15;
  const bf16_t* p = src + (size_t)(k0 + 16 * half) * ld + n0 + n;
  v16bf b;
#pragma unroll
  for (int e = 0; e < 16; ++e) b[e] = p[(size_t)e * ld];
  return b;
}
// B operand that is K^T: (K-dim = feature d, col = key). Per lane: fixed key row, 16 contiguous d.
static __device__ __forceinline__ v16bf load_kT_bf16(const bf16_t* src, int keyrow0, int d0, int ld, int lane) {
  int half = lane >> 4, n = lane & 15;
  const bf16_t* p = src + (size_t)(keyrow0 + n) * ld + d0 + 16 * half;
  v8bf lo = *(const v8bf*)p;
  v8bf hi = *(const v8bf*)(p + 8);
  v16bf b;
#pragma unroll
  for (int e = 0; e < 8; ++e) { b[e] = lo[e]; b[8 + e] = hi[e]; }
  return b;
}
// B operand V: (K-dim = key, col = feature d). Per lane: fixed col d, strided over key rows.
static __device__ __forceinline__ v16bf load_v_bf16(const bf16_t* src, int keyrow0, int d0, int ld, int lane) {
  int half = lane >> 4, n = lane & 15;
  const bf16_t* p = src + (size_t)(keyrow0 + 16 * half) * ld + d0 + n;
  v16bf b;
#pragma unroll
  for (int e = 0; e < 16; ++e) b[e] = p[(size_t)e * ld];
  return b;
}

// reductions across the 16 lanes of one half-wave (rows of one C/D half)
static __device__ __forceinline__ float redmax16(float v) {
#pragma unroll
  for (int m = 1; m < 16; m <<= 1) v = fmaxf(v, __shfl_xor(v, m, 32));
  return v;
}
static __device__ __forceinline__ float redsum16(float v) {
#pragma unroll
  for (int m = 1; m < 16; m <<= 1) v += __shfl_xor(v, m, 32);
  return v;
}

#if HAVE_TDM
// 2-D Tensor Data Mover load: tile_w x tile_h elements, row stride `stride_elems`,
// element size 2^ds_code bytes. D# layout per CDNA5 ISA.
static __device__ __forceinline__ void tdm_load_2d(unsigned int lds_addr, const void* gptr,
                                                   unsigned int tile_w, unsigned int tile_h,
                                                   unsigned int stride_elems, unsigned int ds_code) {
  unsigned long long ga = (unsigned long long)gptr;
  u32x4 g0;
  g0[0] = 1u;                                                     // count=1 (valid user D#)
  g0[1] = lds_addr;                                               // lds_addr [63:32]
  g0[2] = (unsigned int)(ga & 0xFFFFFFFFull);                     // global_addr lo
  g0[3] = (unsigned int)((ga >> 32) & 0x1FFFFFFull) | (2u << 30); // global_addr hi | type=2
  i32x8 g1;
  g1[0] = (int)(ds_code << 16);                                   // data_size [17:16]
  g1[1] = (int)((tile_w & 0xFFFFu) << 16);                        // tensor_dim0 lo16 @ [63:48]
  g1[2] = (int)(((tile_w >> 16) & 0xFFFFu) | ((tile_h & 0xFFFFu) << 16)); // dim0 hi | dim1 lo
  g1[3] = (int)(((tile_h >> 16) & 0xFFFFu) | ((tile_w & 0xFFFFu) << 16)); // dim1 hi | tile_dim0
  g1[4] = (int)(tile_h & 0xFFFFu);                                // tile_dim1 (tile_dim2=0)
  g1[5] = (int)stride_elems;                                      // tensor_dim0_stride lo32
  g1[6] = 0;
  g1[7] = 0;
  i32x4 z4 = {0, 0, 0, 0};
#if __clang_major__ >= 23
  i32x8 z8 = {0, 0, 0, 0, 0, 0, 0, 0};
  __builtin_amdgcn_tensor_load_to_lds(g0, g1, z4, z4, z8, 0);
#else
  __builtin_amdgcn_tensor_load_to_lds(g0, g1, z4, z4, 0);
#endif
}
#define LDS_ADDR(p) ((unsigned int)(size_t)(const void*)(p))
#define MEM_FENCE() asm volatile("" ::: "memory")
#endif

// ---------------- kernel 0: one-time f32 -> bf16 weight conversion ----------------
__global__ void cvt_bf16_kernel(const float* __restrict__ src, bf16_t* __restrict__ dst, int n4) {
  int i = blockIdx.x * blockDim.x + threadIdx.x;
  if (i < n4) {
    v4f x = *(const v4f*)(src + (size_t)i * 4);
    v4bf y;
#pragma unroll
    for (int t = 0; t < 4; ++t) y[t] = (bf16_t)x[t];
    *(v4bf*)(dst + (size_t)i * 4) = y;
  }
}

// ---------------- kernel 1: QKV = X @ W_qkv  (X f32, W pre-converted bf16) ----------------
__global__ void qkv_gemm_kernel(const float* __restrict__ X, const bf16_t* __restrict__ W,
                                bf16_t* __restrict__ QKV) {
  int lane = threadIdx.x;
  int n0 = blockIdx.x * 64;
  int m0 = blockIdx.y * 16;
  v8f acc[4] = {};
#if HAVE_TDM
  __shared__ __align__(16) bf16_t wTile[2][32 * 64]; // double-buffered 32K x 64N bf16 tiles
  tdm_load_2d(LDS_ADDR(wTile[0]), W + n0, 64u, 32u, (unsigned)QKVN, 1u);
#endif
  for (int k0 = 0; k0 < DMODEL; k0 += 32) {
    if (k0 + 32 < DMODEL)
      __builtin_prefetch(X + (size_t)(m0 + (lane & 15)) * DMODEL + k0 + 32, 0, 0);
    v16bf a = load_a_f32(X, m0, k0, DMODEL, lane);
#if HAVE_TDM
    int cur = (k0 >> 5) & 1;
    MEM_FENCE();
    if (k0 + 32 < DMODEL) {
      tdm_load_2d(LDS_ADDR(wTile[cur ^ 1]), W + (size_t)(k0 + 32) * QKVN + n0, 64u, 32u, (unsigned)QKVN, 1u);
      __builtin_amdgcn_s_wait_tensorcnt(1); // current tile resident (in-order TDM)
    } else {
      __builtin_amdgcn_s_wait_tensorcnt(0);
    }
    MEM_FENCE();
    const bf16_t* Wt = wTile[cur];
#pragma unroll
    for (int c = 0; c < 4; ++c) {
      v16bf b = load_b_bf16(Wt, 0, 16 * c, 64, lane);
      acc[c] = wmma_bf16(a, b, acc[c]);
    }
#else
#pragma unroll
    for (int c = 0; c < 4; ++c) {
      v16bf b = load_b_bf16(W, k0, n0 + 16 * c, QKVN, lane);
      acc[c] = wmma_bf16(a, b, acc[c]);
    }
#endif
  }
  int half = lane >> 4, col = lane & 15;
#pragma unroll
  for (int c = 0; c < 4; ++c)
#pragma unroll
    for (int j = 0; j < 8; ++j)
      QKV[(size_t)(m0 + 8 * half + j) * QKVN + n0 + 16 * c + col] = (bf16_t)acc[c][j];
}

// ---------------- kernel 2: differential flash attention ----------------
__global__ void diff_attn_kernel(const bf16_t* __restrict__ QKV, const float* __restrict__ lam,
                                 bf16_t* __restrict__ AO) {
  __shared__ __align__(16) bf16_t P1s[16 * 32];
  __shared__ __align__(16) bf16_t P2s[16 * 32];
#if HAVE_TDM
  __shared__ __align__(16) bf16_t kvTile[2][32 * 192]; // 32 keys x (K1|K2|V), double-buffered
#endif
  int lane = threadIdx.x;
  int i0 = blockIdx.x * 16;
  int h  = blockIdx.y;
  int b  = blockIdx.z;
  int half = lane >> 4, col = lane & 15;
  int rowb = b * TT;
  int hbase = h * (5 * HD);
  float lam_c = fminf(fmaxf(lam[h], 0.f), 1.f);
  int nsteps = ((i0 + 15) >> 5) + 1; // j0 = 32*s, while j0 <= i0+15

#if HAVE_TDM
  tdm_load_2d(LDS_ADDR(kvTile[0]), QKV + (size_t)rowb * QKVN + hbase + 2 * HD,
              192u, 32u, (unsigned)QKVN, 1u);
#endif

  // Q fragments (two 32-wide K chunks covering d=0..63)
  v16bf q1f[2], q2f[2];
#pragma unroll
  for (int dc = 0; dc < 2; ++dc) {
    q1f[dc] = load_a_bf16(QKV, rowb + i0, hbase + 0 * HD + dc * 32, QKVN, lane);
    q2f[dc] = load_a_bf16(QKV, rowb + i0, hbase + 1 * HD + dc * 32, QKVN, lane);
  }

  float m1[8], l1[8], m2[8], l2[8], c1[8], c2[8];
#pragma unroll
  for (int j = 0; j < 8; ++j) { m1[j] = -INFINITY; m2[j] = -INFINITY; l1[j] = 0.f; l2[j] = 0.f; }
  v8f o1[4] = {}, o2[4] = {};

  for (int s = 0; s < nsteps; ++s) {
    int j0 = s << 5;
#if HAVE_TDM
    int cur = s & 1;
    MEM_FENCE();
    if (s + 1 < nsteps) {
      tdm_load_2d(LDS_ADDR(kvTile[cur ^ 1]),
                  QKV + (size_t)(rowb + j0 + 32) * QKVN + hbase + 2 * HD,
                  192u, 32u, (unsigned)QKVN, 1u);
      __builtin_amdgcn_s_wait_tensorcnt(1);
    } else {
      __builtin_amdgcn_s_wait_tensorcnt(0);
    }
    MEM_FENCE();
    const bf16_t* Ksrc = kvTile[cur];
    int krow0 = 0, kld = 192, kbase = 0;
#else
    const bf16_t* Ksrc = QKV;
    int krow0 = rowb + j0, kld = QKVN, kbase = hbase + 2 * HD;
#endif

    v8f s1[2] = {}, s2[2] = {};
#pragma unroll
    for (int sub = 0; sub < 2; ++sub)
#pragma unroll
      for (int dc = 0; dc < 2; ++dc) {
        v16bf bk1 = load_kT_bf16(Ksrc, krow0 + 16 * sub, kbase + 0 * HD + dc * 32, kld, lane);
        s1[sub] = wmma_bf16(q1f[dc], bk1, s1[sub]);
        v16bf bk2 = load_kT_bf16(Ksrc, krow0 + 16 * sub, kbase + 1 * HD + dc * 32, kld, lane);
        s2[sub] = wmma_bf16(q2f[dc], bk2, s2[sub]);
      }

    // per-row online softmax update for both streams; P tiles staged to LDS as bf16
#pragma unroll
    for (int j = 0; j < 8; ++j) {
      int row_g = i0 + 8 * half + j;
      // stream 1
      float a0 = s1[0][j] * 0.125f + ((j0 + col)      <= row_g ? 0.f : -1e9f);
      float a1 = s1[1][j] * 0.125f + ((j0 + 16 + col) <= row_g ? 0.f : -1e9f);
      float tmax = redmax16(fmaxf(a0, a1));
      float mn = fmaxf(m1[j], tmax);
      float cc = __expf(m1[j] - mn);
      float p0 = __expf(a0 - mn);
      float p1 = __expf(a1 - mn);
      l1[j] = l1[j] * cc + redsum16(p0 + p1);
      m1[j] = mn; c1[j] = cc;
      P1s[(8 * half + j) * 32 + col]      = (bf16_t)p0;
      P1s[(8 * half + j) * 32 + 16 + col] = (bf16_t)p1;
      // stream 2
      float b0 = s2[0][j] * 0.125f + ((j0 + col)      <= row_g ? 0.f : -1e9f);
      float b1 = s2[1][j] * 0.125f + ((j0 + 16 + col) <= row_g ? 0.f : -1e9f);
      float tmax2 = redmax16(fmaxf(b0, b1));
      float mn2 = fmaxf(m2[j], tmax2);
      float cc2 = __expf(m2[j] - mn2);
      float q0 = __expf(b0 - mn2);
      float q1 = __expf(b1 - mn2);
      l2[j] = l2[j] * cc2 + redsum16(q0 + q1);
      m2[j] = mn2; c2[j] = cc2;
      P2s[(8 * half + j) * 32 + col]      = (bf16_t)q0;
      P2s[(8 * half + j) * 32 + 16 + col] = (bf16_t)q1;
    }

    // rescale accumulators
#pragma unroll
    for (int vd = 0; vd < 4; ++vd)
#pragma unroll
      for (int j = 0; j < 8; ++j) { o1[vd][j] *= c1[j]; o2[vd][j] *= c2[j]; }

    __syncthreads();
    v16bf pa1 = load_a_bf16(P1s, 0, 0, 32, lane);
    v16bf pa2 = load_a_bf16(P2s, 0, 0, 32, lane);
    __syncthreads();

#pragma unroll
    for (int vd = 0; vd < 4; ++vd) {
      v16bf bv = load_v_bf16(Ksrc, krow0, kbase + 2 * HD + vd * 16, kld, lane);
      o1[vd] = wmma_bf16(pa1, bv, o1[vd]);
      o2[vd] = wmma_bf16(pa2, bv, o2[vd]);
    }
  }

#pragma unroll
  for (int vd = 0; vd < 4; ++vd)
#pragma unroll
    for (int j = 0; j < 8; ++j) {
      float r = o1[vd][j] / l1[j] - lam_c * (o2[vd][j] / l2[j]);
      AO[(size_t)(rowb + i0 + 8 * half + j) * AON + h * HD + vd * 16 + col] = (bf16_t)r;
    }
}

// ---------------- kernel 3: OUT = AO @ W_out (both bf16 -> f32) ----------------
__global__ void out_gemm_kernel(const bf16_t* __restrict__ A, const bf16_t* __restrict__ W,
                                float* __restrict__ OUT) {
  int lane = threadIdx.x;
  int n0 = blockIdx.x * 64;
  int m0 = blockIdx.y * 16;
  v8f acc[4] = {};
#if HAVE_TDM
  __shared__ __align__(16) bf16_t wTile[2][32 * 64];
  tdm_load_2d(LDS_ADDR(wTile[0]), W + n0, 64u, 32u, (unsigned)DMODEL, 1u);
#endif
  for (int k0 = 0; k0 < AON; k0 += 32) {
    if (k0 + 32 < AON)
      __builtin_prefetch(A + (size_t)(m0 + (lane & 15)) * AON + k0 + 32, 0, 0);
    v16bf a = load_a_bf16(A, m0, k0, AON, lane);
#if HAVE_TDM
    int cur = (k0 >> 5) & 1;
    MEM_FENCE();
    if (k0 + 32 < AON) {
      tdm_load_2d(LDS_ADDR(wTile[cur ^ 1]), W + (size_t)(k0 + 32) * DMODEL + n0, 64u, 32u, (unsigned)DMODEL, 1u);
      __builtin_amdgcn_s_wait_tensorcnt(1);
    } else {
      __builtin_amdgcn_s_wait_tensorcnt(0);
    }
    MEM_FENCE();
    const bf16_t* Wt = wTile[cur];
#pragma unroll
    for (int c = 0; c < 4; ++c) {
      v16bf b = load_b_bf16(Wt, 0, 16 * c, 64, lane);
      acc[c] = wmma_bf16(a, b, acc[c]);
    }
#else
#pragma unroll
    for (int c = 0; c < 4; ++c) {
      v16bf b = load_b_bf16(W, k0, n0 + 16 * c, DMODEL, lane);
      acc[c] = wmma_bf16(a, b, acc[c]);
    }
#endif
  }
  int half = lane >> 4, col = lane & 15;
#pragma unroll
  for (int c = 0; c < 4; ++c)
#pragma unroll
    for (int j = 0; j < 8; ++j)
      OUT[(size_t)(m0 + 8 * half + j) * DMODEL + n0 + 16 * c + col] = acc[c][j];
}

extern "C" void kernel_launch(void* const* d_in, const int* in_sizes, int n_in,
                              void* d_out, int out_size, void* d_ws, size_t ws_size,
                              hipStream_t stream) {
  const float* x    = (const float*)d_in[0];
  // d_in[1] is the causal mask; reproduced analytically in-kernel.
  const float* Wqkv = (const float*)d_in[2];
  const float* Wout = (const float*)d_in[3];
  const float* lam  = (const float*)d_in[4];
  float* out = (float*)d_out;

  // workspace layout (256B-aligned sections)
  char* ws = (char*)d_ws;
  size_t off = 0;
  bf16_t* qkv = (bf16_t*)(ws + off);
  off += ((size_t)BB * TT * QKVN * sizeof(bf16_t) + 255) & ~(size_t)255;
  bf16_t* ao = (bf16_t*)(ws + off);
  off += ((size_t)BB * TT * AON * sizeof(bf16_t) + 255) & ~(size_t)255;
  bf16_t* wqkv_b = (bf16_t*)(ws + off);
  off += ((size_t)DMODEL * QKVN * sizeof(bf16_t) + 255) & ~(size_t)255;
  bf16_t* wout_b = (bf16_t*)(ws + off);

  // one-time weight conversions (f32 -> bf16)
  {
    int n4 = (DMODEL * QKVN) / 4;
    cvt_bf16_kernel<<<(n4 + 255) / 256, 256, 0, stream>>>(Wqkv, wqkv_b, n4);
    int m4 = (DMODEL * DMODEL) / 4;
    cvt_bf16_kernel<<<(m4 + 255) / 256, 256, 0, stream>>>(Wout, wout_b, m4);
  }

  dim3 g1(QKVN / 64, (BB * TT) / 16);
  qkv_gemm_kernel<<<g1, 32, 0, stream>>>(x, wqkv_b, qkv);

  dim3 g2(TT / 16, NH, BB);
  diff_attn_kernel<<<g2, 32, 0, stream>>>(qkv, lam, ao);

  dim3 g3(DMODEL / 64, (BB * TT) / 16);
  out_gemm_kernel<<<g3, 32, 0, stream>>>(ao, wout_b, out);
}
